// GRUFromScratch_59931973648424
// MI455X (gfx1250) — matrix operand order repaired and checked by reading, not compiled, not run on hardware
//
#include <hip/hip_runtime.h>
#include <hip/hip_bf16.h>
#include <stdint.h>

#define S_LEN 512
#define BATCH 64
#define DIM   1024   // D_IN == D_H == D_OUT

typedef __attribute__((ext_vector_type(16))) __bf16 v16bf;
typedef __attribute__((ext_vector_type(8)))  __bf16 v8bf;
typedef __attribute__((ext_vector_type(8)))  float  v8f;
typedef __attribute__((ext_vector_type(4)))  unsigned int v4u;
typedef __attribute__((ext_vector_type(8)))  int v8i;
typedef __attribute__((ext_vector_type(4)))  int v4i;

// LDS row pitch for staged A-slabs: 32 bf16 (64B) data + 16B TDM pad = 80B.
// 20 DWORDs/row -> fragment reads are LDS-bank-conflict-free.
#define ROW_STRIDE 40                      // elements (bf16) per LDS row
#define ROW_BYTES  (ROW_STRIDE * 2)        // 80

// ---------------------------------------------------------------------------
// Tensor Data Mover: async-copy a (rows x 32) bf16 tile, global row stride
// ldK elements, into LDS at lds_off with 16B pad every 64B (80B LDS pitch).
// Completion tracked by TENSORcnt on the issuing wave.
// ---------------------------------------------------------------------------
__device__ __forceinline__ void tdm_load_slab(unsigned lds_off,
                                              const __bf16* gptr,
                                              int ldK, int rows) {
  unsigned long long ga = (unsigned long long)(uintptr_t)gptr;
  v4u g0;
  g0[0] = 1u;                                    // count=1, user mode
  g0[1] = lds_off;                               // LDS byte address
  g0[2] = (unsigned)ga;                          // global_addr[31:0]
  g0[3] = (unsigned)((ga >> 32) & 0x01FFFFFFu)   // global_addr[56:32]
          | (2u << 30);                          // type = 2 ("image")
  v8i g1;
  g1[0] = (int)((1u << 16) |                     // data_size = 1 (2 bytes)
                (1u << 20) |                     // pad_enable
                (3u << 22) |                     // pad_interval: 16 DWORDs (64B)
                (3u << 25));                     // pad_amount:    4 DWORDs (16B)
  g1[1] = (int)(32u << 16);                      // tensor_dim0 = 32 (lo16)
  g1[2] = (int)((unsigned)rows << 16);           // dim0 hi=0 | tensor_dim1 = rows
  g1[3] = (int)(32u << 16);                      // dim1 hi=0 | tile_dim0 = 32
  g1[4] = rows;                                  // tile_dim1 = rows, tile_dim2 = 0
  g1[5] = ldK;                                   // tensor_dim0_stride[31:0]
  g1[6] = 0;                                     // stride hi | dim1_stride lo
  g1[7] = 0;
  v4i zg = {0, 0, 0, 0};
#if __clang_major__ >= 23
  v8i zg8 = {0, 0, 0, 0, 0, 0, 0, 0};
  __builtin_amdgcn_tensor_load_to_lds(g0, g1, zg, zg, zg8, 0);
#else
  __builtin_amdgcn_tensor_load_to_lds(g0, g1, zg, zg, 0);
#endif
}

// ---------------------------------------------------------------------------
// WMMA fragment loaders (CDNA5 16x16x32 bf16 layouts, wave32)
// ---------------------------------------------------------------------------
__device__ __forceinline__ v16bf lds_fragA(const __bf16* buf, int row0, int lane) {
  // A (16x32): lanes 0-15 hold K0-7 / K16-23 of row M=lane; lanes 16-31 K8-15 / K24-31
  const __bf16* base = buf + (row0 + (lane & 15)) * ROW_STRIDE + ((lane >> 4) << 3);
  union { v16bf v; struct { v8bf lo, hi; } s; } u;
  u.s.lo = *(const v8bf*)(base);        // ds_load_b128
  u.s.hi = *(const v8bf*)(base + 16);
  return u.v;
}

__device__ __forceinline__ v16bf load_fragA_g(const __bf16* __restrict__ p, int ld,
                                              int row0, int k0, int lane) {
  const __bf16* base = p + (size_t)(row0 + (lane & 15)) * ld + (k0 + ((lane >> 4) << 3));
  union { v16bf v; struct { v8bf lo, hi; } s; } u;
  u.s.lo = *(const v8bf*)(base);
  u.s.hi = *(const v8bf*)(base + 16);
  return u.v;
}

__device__ __forceinline__ v16bf load_fragB(const __bf16* __restrict__ p, int ld,
                                            int n0, int k0, int lane) {
  // B (32x16) = W^T: lanes 0-15 col N=lane hold K0-7 / K16-23; lanes 16-31 K8-15 / K24-31
  const __bf16* base = p + (size_t)(n0 + (lane & 15)) * ld + (k0 + ((lane >> 4) << 3));
  union { v16bf v; struct { v8bf lo, hi; } s; } u;
  u.s.lo = *(const v8bf*)(base);
  u.s.hi = *(const v8bf*)(base + 16);
  return u.v;
}

__device__ __forceinline__ v8f wmma_bf16(v16bf a, v16bf b, v8f c) {
  return __builtin_amdgcn_wmma_f32_16x16x32_bf16(false, a, false, b, (short)0, c,
                                                 false, false);
}

// ---------------------------------------------------------------------------
// Big GEMM: out[M,N] = A[M,K](bf16) @ W[N,K]^T (bf16) + bias.
// WG tile 128(M) x 64(N); 8 waves, each 4 M-tiles x 1 N-tile.
// A slab (128x32) TDM-staged into double-buffered LDS; W direct + prefetch.
// ---------------------------------------------------------------------------
#define GEMM_BUF_ELEMS (128 * ROW_STRIDE)
#define GEMM_BUF_BYTES (GEMM_BUF_ELEMS * 2)

template <bool OUT_F32>
__global__ __launch_bounds__(256) void gemm_bf16(
    const __bf16* __restrict__ A, const __bf16* __restrict__ W,
    const float* __restrict__ bias,
    float* __restrict__ outF, __bf16* __restrict__ outB,
    int M, int N, int K) {
  __shared__ __attribute__((aligned(64))) __bf16 smemA[2 * GEMM_BUF_ELEMS];
  const int lane = threadIdx.x & 31;
  const int wave = threadIdx.x >> 5;
  const int mrel = (wave & 1) * 64;
  const int m0   = blockIdx.y * 128 + mrel;
  const int n0   = blockIdx.x * 64 + (wave >> 1) * 16;
  (void)M;
  const __bf16* Aslab = A + (size_t)blockIdx.y * 128 * K;
  const unsigned lds_base = (unsigned)(uintptr_t)&smemA[0];

  if (wave == 0) tdm_load_slab(lds_base, Aslab, K, 128);

  v8f acc[4] = {};
  int bufsel = 0;
  for (int k = 0; k < K; k += 32) {
    __syncthreads();                       // buf[bufsel^1] free for overwrite
    if (wave == 0) {
      if (k + 32 < K) {
        tdm_load_slab(lds_base + (unsigned)((bufsel ^ 1) * GEMM_BUF_BYTES),
                      Aslab + (k + 32), K, 128);
        __builtin_amdgcn_s_wait_tensorcnt((short)1);   // current slab done
      } else {
        __builtin_amdgcn_s_wait_tensorcnt((short)0);
      }
    }
    __syncthreads();                       // slab visible to all waves
    if (k + 32 < K)
      __builtin_prefetch(W + (size_t)(n0 + (lane & 15)) * K + k + 32, 0, 1);
    v16bf bfr = load_fragB(W, K, n0, k, lane);
    const __bf16* bufp = smemA + bufsel * GEMM_BUF_ELEMS;
#pragma unroll
    for (int i = 0; i < 4; ++i) {
      v16bf afr = lds_fragA(bufp, mrel + 16 * i, lane);
      acc[i] = wmma_bf16(afr, bfr, acc[i]);
    }
    bufsel ^= 1;
  }

  const int col  = n0 + (lane & 15);
  const int rsub = (lane >> 4) << 3;
  const float bv = bias ? bias[col] : 0.0f;
#pragma unroll
  for (int i = 0; i < 4; ++i) {
#pragma unroll
    for (int v = 0; v < 8; ++v) {
      size_t idx = (size_t)(m0 + 16 * i + rsub + v) * N + col;
      float val  = acc[i][v] + bv;
      if (OUT_F32) outF[idx] = val;
      else         outB[idx] = (__bf16)val;
    }
  }
}

// ---------------------------------------------------------------------------
// GRU step, phase A: z = sigmoid(pz_t + h@Whz^T), r = sigmoid(pr_t + h@Whr^T)
// 16 WGs x 8 waves; waves 0-63 -> z, 64-127 -> r. h slab TDM-staged in LDS.
// ---------------------------------------------------------------------------
#define STEP_BUF_ELEMS (64 * ROW_STRIDE)
#define STEP_BUF_BYTES (STEP_BUF_ELEMS * 2)

__global__ __launch_bounds__(256) void gru_step_zr(
    const __bf16* __restrict__ h_bf, const float* __restrict__ h_f32,
    const __bf16* __restrict__ Whz, const __bf16* __restrict__ Whr,
    const __bf16* __restrict__ pz_t, const __bf16* __restrict__ pr_t,
    float* __restrict__ z_out, __bf16* __restrict__ rh_out) {
  __shared__ __attribute__((aligned(64))) __bf16 smemH[2 * STEP_BUF_ELEMS];
  const int lane = threadIdx.x & 31;
  const int wave = threadIdx.x >> 5;
  const int gw   = blockIdx.x * 8 + wave;   // 0..127
  const int gate = gw >> 6;                 // 0 = z, 1 = r
  const int n0   = (gw & 63) * 16;
  const __bf16* Wg = gate ? Whr : Whz;
  const unsigned lds_base = (unsigned)(uintptr_t)&smemH[0];

  if (wave == 0) tdm_load_slab(lds_base, h_bf, DIM, 64);

  v8f acc[4] = {};
  int bufsel = 0;
  for (int k = 0; k < DIM; k += 32) {
    __syncthreads();
    if (wave == 0) {
      if (k + 32 < DIM) {
        tdm_load_slab(lds_base + (unsigned)((bufsel ^ 1) * STEP_BUF_BYTES),
                      h_bf + (k + 32), DIM, 64);
        __builtin_amdgcn_s_wait_tensorcnt((short)1);
      } else {
        __builtin_amdgcn_s_wait_tensorcnt((short)0);
      }
    }
    __syncthreads();
    v16bf bfr = load_fragB(Wg, DIM, n0, k, lane);
    const __bf16* bufp = smemH + bufsel * STEP_BUF_ELEMS;
#pragma unroll
    for (int i = 0; i < 4; ++i) {
      v16bf afr = lds_fragA(bufp, 16 * i, lane);
      acc[i] = wmma_bf16(afr, bfr, acc[i]);
    }
    bufsel ^= 1;
  }

  const int col  = n0 + (lane & 15);
  const int rsub = (lane >> 4) << 3;
#pragma unroll
  for (int i = 0; i < 4; ++i) {
#pragma unroll
    for (int v = 0; v < 8; ++v) {
      int row = 16 * i + rsub + v;
      int idx = row * DIM + col;
      if (gate == 0) {
        float g = acc[i][v] + (float)pz_t[idx];
        z_out[idx] = 1.0f / (1.0f + __expf(-g));
      } else {
        float g = acc[i][v] + (float)pr_t[idx];
        float r = 1.0f / (1.0f + __expf(-g));
        rh_out[idx] = (__bf16)(r * h_f32[idx]);
      }
    }
  }
}

// ---------------------------------------------------------------------------
// GRU step, phase B: n = tanh(pn_t + (r*h)@Whn^T); h = (1-z)*h + z*n
// 8 WGs x 8 waves. r*h slab TDM-staged in LDS.
// ---------------------------------------------------------------------------
__global__ __launch_bounds__(256) void gru_step_n(
    const __bf16* __restrict__ rh_bf, const __bf16* __restrict__ Whn,
    const __bf16* __restrict__ pn_t, const float* __restrict__ z_in,
    float* __restrict__ h_f32, __bf16* __restrict__ h_bf,
    __bf16* __restrict__ hs_t, float* __restrict__ h_last) {
  __shared__ __attribute__((aligned(64))) __bf16 smemH[2 * STEP_BUF_ELEMS];
  const int lane = threadIdx.x & 31;
  const int wave = threadIdx.x >> 5;
  const int gw   = blockIdx.x * 8 + wave;   // 0..63
  const int n0   = gw * 16;
  const unsigned lds_base = (unsigned)(uintptr_t)&smemH[0];

  if (wave == 0) tdm_load_slab(lds_base, rh_bf, DIM, 64);

  v8f acc[4] = {};
  int bufsel = 0;
  for (int k = 0; k < DIM; k += 32) {
    __syncthreads();
    if (wave == 0) {
      if (k + 32 < DIM) {
        tdm_load_slab(lds_base + (unsigned)((bufsel ^ 1) * STEP_BUF_BYTES),
                      rh_bf + (k + 32), DIM, 64);
        __builtin_amdgcn_s_wait_tensorcnt((short)1);
      } else {
        __builtin_amdgcn_s_wait_tensorcnt((short)0);
      }
    }
    __syncthreads();
    v16bf bfr = load_fragB(Whn, DIM, n0, k, lane);
    const __bf16* bufp = smemH + bufsel * STEP_BUF_ELEMS;
#pragma unroll
    for (int i = 0; i < 4; ++i) {
      v16bf afr = lds_fragA(bufp, 16 * i, lane);
      acc[i] = wmma_bf16(afr, bfr, acc[i]);
    }
    bufsel ^= 1;
  }

  const int col  = n0 + (lane & 15);
  const int rsub = (lane >> 4) << 3;
#pragma unroll
  for (int i = 0; i < 4; ++i) {
#pragma unroll
    for (int v = 0; v < 8; ++v) {
      int row = 16 * i + rsub + v;
      int idx = row * DIM + col;
      float n  = tanhf(acc[i][v] + (float)pn_t[idx]);
      float z  = z_in[idx];
      float hn = (1.0f - z) * h_f32[idx] + z * n;
      h_f32[idx] = hn;              // each element owned by exactly one lane
      h_bf[idx]  = (__bf16)hn;
      hs_t[idx]  = (__bf16)hn;
      if (h_last) h_last[idx] = hn;
    }
  }
}

// ---------------------------------------------------------------------------
// Helpers: f32 -> bf16 conversion, h init
// ---------------------------------------------------------------------------
__global__ void cvt_f32_to_bf16(const float* __restrict__ s,
                                __bf16* __restrict__ d, long n) {
  long i = (long)blockIdx.x * blockDim.x + threadIdx.x;
  if (i < n) d[i] = (__bf16)s[i];
}

__global__ void init_h(const float* __restrict__ h0, float* __restrict__ hf,
                       __bf16* __restrict__ hb, int n) {
  int i = blockIdx.x * blockDim.x + threadIdx.x;
  if (i < n) { float v = h0[i]; hf[i] = v; hb[i] = (__bf16)v; }
}

// ---------------------------------------------------------------------------
extern "C" void kernel_launch(void* const* d_in, const int* in_sizes, int n_in,
                              void* d_out, int out_size, void* d_ws, size_t ws_size,
                              hipStream_t stream) {
  (void)in_sizes; (void)n_in; (void)out_size; (void)ws_size;
  const float* x_seq = (const float*)d_in[0];
  const float* h0    = (const float*)d_in[1];
  const float* Wxz_w = (const float*)d_in[2];
  const float* Wxz_b = (const float*)d_in[3];
  const float* Whz_w = (const float*)d_in[4];
  const float* Wxr_w = (const float*)d_in[5];
  const float* Wxr_b = (const float*)d_in[6];
  const float* Whr_w = (const float*)d_in[7];
  const float* Wxn_w = (const float*)d_in[8];
  const float* Wxn_b = (const float*)d_in[9];
  const float* Whn_w = (const float*)d_in[10];
  const float* Why_w = (const float*)d_in[11];
  const float* Why_b = (const float*)d_in[12];

  const size_t SB = (size_t)S_LEN * BATCH;   // 32768 rows
  const size_t MK = SB * DIM;                // x / proj / hs element count
  const size_t WD = (size_t)DIM * DIM;

  char* ws = (char*)d_ws;
  auto take = [&ws](size_t bytes) {
    char* p = ws; ws += (bytes + 255) & ~(size_t)255; return p;
  };
  __bf16* Wz   = (__bf16*)take(WD * 2);
  __bf16* Wr   = (__bf16*)take(WD * 2);
  __bf16* Wn   = (__bf16*)take(WD * 2);
  __bf16* Uz   = (__bf16*)take(WD * 2);
  __bf16* Ur   = (__bf16*)take(WD * 2);
  __bf16* Un   = (__bf16*)take(WD * 2);
  __bf16* Wy   = (__bf16*)take(WD * 2);
  __bf16* x_bf = (__bf16*)take(MK * 2);
  __bf16* pz   = (__bf16*)take(MK * 2);
  __bf16* pr   = (__bf16*)take(MK * 2);
  __bf16* pn   = (__bf16*)take(MK * 2);
  __bf16* hs   = (__bf16*)take(MK * 2);
  float*  h_f  = (float*) take((size_t)BATCH * DIM * 4);
  __bf16* h_b  = (__bf16*)take((size_t)BATCH * DIM * 2);
  __bf16* rh   = (__bf16*)take((size_t)BATCH * DIM * 2);
  float*  z_f  = (float*) take((size_t)BATCH * DIM * 4);

  auto cvt = [&](const float* src, __bf16* dst, size_t n) {
    cvt_f32_to_bf16<<<(unsigned)((n + 255) / 256), 256, 0, stream>>>(src, dst, (long)n);
  };
  cvt(Wxz_w, Wz, WD); cvt(Wxr_w, Wr, WD); cvt(Wxn_w, Wn, WD);
  cvt(Whz_w, Uz, WD); cvt(Whr_w, Ur, WD); cvt(Whn_w, Un, WD);
  cvt(Why_w, Wy, WD); cvt(x_seq, x_bf, MK);
  init_h<<<(BATCH * DIM + 255) / 256, 256, 0, stream>>>(h0, h_f, h_b, BATCH * DIM);

  // Parallel input projections: one big WMMA GEMM per gate (M=32768,N=K=1024)
  dim3 gBig(DIM / 64, (unsigned)(SB / 128));
  gemm_bf16<false><<<gBig, 256, 0, stream>>>(x_bf, Wz, Wxz_b, nullptr, pz, (int)SB, DIM, DIM);
  gemm_bf16<false><<<gBig, 256, 0, stream>>>(x_bf, Wr, Wxr_b, nullptr, pr, (int)SB, DIM, DIM);
  gemm_bf16<false><<<gBig, 256, 0, stream>>>(x_bf, Wn, Wxn_b, nullptr, pn, (int)SB, DIM, DIM);

  // Serial scan: stream ordering provides the inter-step dependency
  float* h_last = (float*)d_out + MK;      // d_out = outputs (SB*DIM) ++ h_last
  const size_t stepE = (size_t)BATCH * DIM;
  for (int t = 0; t < S_LEN; ++t) {
    gru_step_zr<<<16, 256, 0, stream>>>(h_b, h_f, Uz, Ur,
                                        pz + (size_t)t * stepE, pr + (size_t)t * stepE,
                                        z_f, rh);
    gru_step_n<<<8, 256, 0, stream>>>(rh, Un, pn + (size_t)t * stepE, z_f,
                                      h_f, h_b, hs + (size_t)t * stepE,
                                      (t == S_LEN - 1) ? h_last : nullptr);
  }

  // Output projection (f32 out straight to d_out)
  gemm_bf16<true><<<gBig, 256, 0, stream>>>(hs, Wy, Why_b, (float*)d_out, nullptr,
                                            (int)SB, DIM, DIM);
}
